// LocalPPM_34969623724317
// MI455X (gfx1250) — compile-verified, matrix operand order checked by reading, not run on hardware
//
#include <hip/hip_runtime.h>
#include <hip/hip_bf16.h>
#include <math.h>

// CDNA5 / gfx1250 wave32 WMMA types
typedef __attribute__((ext_vector_type(16))) __bf16 v16bf;
typedef __attribute__((ext_vector_type(8)))  float  v8f;

#define BB 4
#define CC 256
#define HH 56
#define WW 56
#define RR 2
#define KS 5
#define K2 25
#define TOPK 10
#define TAUINV 10.0f     // 1/0.1
#define EPSV 1e-8f
#define TW 16            // spatial tile width (matches WMMA M/N)
#define HALO 20          // TW + 2*RR
#define NPOS (KS*HALO)   // 100 halo positions
#define KPAD 128         // halo positions padded to 4 WMMA K-steps
#define NTH 128          // 4 wave32 waves
#define NWAVE (NTH/32)

union Frag { v16bf v; unsigned int u[8]; };

// Load one 16-bit A/B fragment: per lane, 8 packed K-pairs at kof[] offsets.
// Offsets form two contiguous 16B runs per lane-half -> lowers to 2x ds_load_b128.
__device__ __forceinline__ void loadFrag(Frag& f, const __bf16* base, const int* kof) {
#pragma unroll
    for (int v = 0; v < 8; ++v)
        f.u[v] = *(const unsigned int*)&base[kof[v]];
}

__device__ __forceinline__ v8f wmma_bf16(const Frag& A, const Frag& B, v8f acc) {
    return __builtin_amdgcn_wmma_f32_16x16x32_bf16(
               false, A.v, false, B.v, (short)0, acc, false, false);
}

__global__ __launch_bounds__(NTH)
void localppm_kernel(const float* __restrict__ x,
                     const float* __restrict__ gammap,
                     float* __restrict__ out)
{
    // [row][col][channel]: K-pairs contiguous in channel (phase-3 Gram)
    __shared__ __bf16 xs[NPOS * CC];            // 51200 B
    // [channel][kpos]: K-pairs contiguous in halo position (phase-5 GEMM)
    __shared__ __bf16 xsT[CC * KPAD];           // 65536 B
    // [outcol][kpos]: banded softmax weights, bf16 (phase-5 B matrix)
    __shared__ __bf16 wbandT[TW * KPAD];        // 4096 B
    __shared__ float  nrm2[NPOS];               // squared norms per halo position
    __shared__ float  dotL[K2 * TW];            // banded Gram: dot[k=(dy,dd)][m]
    __shared__ float  wL[K2 * TW];              // softmax weights

    const int tid  = threadIdx.x;
    const int lane = tid & 31;
    const int wave = tid >> 5;
    const int wt = blockIdx.x;   // w tile
    const int h  = blockIdx.y;
    const int b  = blockIdx.z;
    const int w0 = wt * TW;
    const float gamma = gammap[0];

    // Prefetch next tile's center row into L2 (global_prefetch_b8)
    {
        int gwn = w0 + TW;
        if (gwn < WW)
            __builtin_prefetch(&x[(((size_t)b * CC) * HH + h) * WW + gwn], 0, 1);
    }

    // ---- Phase 1: stage halo (zero padded) into LDS as bf16, both layouts ----
    if (tid < NPOS) nrm2[tid] = 0.f;
    for (int idx = tid; idx < NPOS * CC; idx += NTH) {
        int c   = idx / NPOS;
        int pc  = idx - c * NPOS;
        int row = pc / HALO;
        int col = pc - row * HALO;
        int gh = h + row - RR;
        int gw = w0 + col - RR;
        float v = 0.0f;
        if (gh >= 0 && gh < HH && gw >= 0 && gw < WW)
            v = x[(((size_t)b * CC + c) * HH + gh) * WW + gw];
        __bf16 bv = (__bf16)v;
        xs[pc * CC + c]    = bv;
        xsT[c * KPAD + pc] = bv;
    }
    // zero-pad xsT K-tail (kpos 100..127)
    for (int i = tid; i < CC * (KPAD - NPOS); i += NTH) {
        int c = i / (KPAD - NPOS);
        int k = NPOS + (i - c * (KPAD - NPOS));
        xsT[c * KPAD + k] = (__bf16)0.f;
    }
    __syncthreads();

    // ---- Phase 2: squared norms, 4-way split per position + ds_add_f32 ----
    for (int job = tid; job < NPOS * 4; job += NTH) {
        int pos = job >> 2, q = job & 3;
        const __bf16* p = &xs[pos * CC + q * 64];
        float s = 0.f;
        for (int c = 0; c < 64; ++c) { float v = (float)p[c]; s += v * v; }
        atomicAdd(&nrm2[pos], s);
    }

    // ---- Phase 3: banded Gram, one dy per job, two tiles per job ----
    // Tiles share A (center) fragments; two independent WMMA chains per K-step
    // hide each other's ds_load latency.
    const int mn = lane & 15;   // M for A-rows / N for B-cols
    const int g  = lane >> 4;   // lane half selects K sub-block
    int kof[8];
#pragma unroll
    for (int v = 0; v < 8; ++v)
        kof[v] = (v < 4) ? (8 * g + 2 * v) : (16 + 8 * g + 2 * (v - 4));

    for (int dy = wave; dy < KS; dy += NWAVE) {
        const __bf16* aptr = &xs[(RR * HALO + (mn + RR)) * CC];  // center col m+2
        const __bf16* b0   = &xs[(dy * HALO + mn) * CC];         // tile0: cols 0..15
        const __bf16* b1   = &xs[(dy * HALO + (mn + 4)) * CC];   // tile1: cols 4..19
        v8f acc0 = {}, acc1 = {};
#pragma unroll
        for (int kc = 0; kc < CC; kc += 32) {
            Frag A, B0, B1;
            loadFrag(A,  aptr + kc, kof);
            loadFrag(B0, b0 + kc, kof);
            loadFrag(B1, b1 + kc, kof);
            acc0 = wmma_bf16(A, B0, acc0);   // independent of acc1 chain
            acc1 = wmma_bf16(A, B1, acc1);
        }
        // Scatter in-band Gram entries. D: VGPR r -> M=r+8g; N=lane&15.
#pragma unroll
        for (int r = 0; r < 8; ++r) {
            int mm = r + 8 * g;
            // tile0: col = mn (always < 16)
            int dd0 = mn - mm;
            if (dd0 >= 0 && dd0 <= 4)
                dotL[(dy * KS + dd0) * TW + mm] = acc0[r];
            // tile1: col = mn + 4; dedup -> only cols >= 16
            int col1 = mn + 4;
            int dd1 = col1 - mm;
            if (col1 >= 16 && dd1 >= 0 && dd1 <= 4)
                dotL[(dy * KS + dd1) * TW + mm] = acc1[r];
        }
    }
    __syncthreads();

    // ---- Phase 4: per-position top-10 + masked softmax (16 lanes) ----
    if (tid < TW) {
        const int m = tid;
        float cn = sqrtf(nrm2[RR * HALO + m + RR]);
        float s[K2];
        bool  keep[K2];
        for (int k = 0; k < K2; ++k) {
            int dy = k / KS, dd = k - dy * KS;
            float pn  = sqrtf(nrm2[dy * HALO + m + dd]);
            float den = fmaxf(pn * cn, EPSV);
            s[k] = dotL[k * TW + m] / den * TAUINV;
            keep[k] = false;
        }
        float maxv = -1e30f;
        for (int it = 0; it < TOPK; ++it) {
            int bi = 0; float bv = -1e30f;
            for (int k = 0; k < K2; ++k)
                if (!keep[k] && s[k] > bv) { bv = s[k]; bi = k; }
            keep[bi] = true;
            if (it == 0) maxv = bv;
        }
        float sum = 0.f, e[K2];
        for (int k = 0; k < K2; ++k) {
            e[k] = keep[k] ? __expf(s[k] - maxv) : 0.f;
            sum += e[k];
        }
        float inv = 1.0f / sum;
        for (int k = 0; k < K2; ++k) wL[k * TW + m] = e[k] * inv;
    }
    __syncthreads();

    // ---- Phase 5a: build banded weight matrix WbandT[m][kpos] in bf16 ----
    for (int i = tid; i < TW * KPAD; i += NTH) {
        int m = i >> 7;        // output column 0..15
        int k = i & (KPAD-1);  // kpos
        float v = 0.f;
        if (k < NPOS) {
            int dy = k / HALO, col = k - dy * HALO;
            int dd = col - m;
            if (dd >= 0 && dd <= KS - 1) v = wL[(dy * KS + dd) * TW + m];
        }
        wbandT[m * KPAD + k] = (__bf16)v;
    }
    __syncthreads();

    // ---- Phase 5b: Y[c,m] = XsT[c,:] . WbandT[m,:] via WMMA + fp32 residual ----
    // B fragments depend only on lane column -> hoist all 4 K-steps once.
    // Two channel chunks per iteration share B; two independent WMMA chains.
    Frag WB[4];
#pragma unroll
    for (int kc4 = 0; kc4 < 4; ++kc4)
        loadFrag(WB[kc4], &wbandT[mn * KPAD + 32 * kc4], kof);

    const int gw = w0 + mn;
    for (int pair = wave; pair < (CC / TW) / 2; pair += NWAVE) {
        const int c0 = pair * 32;                       // chunks c0 and c0+16
        const __bf16* a0 = &xsT[(c0 + mn) * KPAD];
        const __bf16* a1 = &xsT[(c0 + 16 + mn) * KPAD];
        v8f acc0 = {}, acc1 = {};
#pragma unroll
        for (int kc4 = 0; kc4 < 4; ++kc4) {
            Frag A0, A1;
            loadFrag(A0, a0 + 32 * kc4, kof);
            loadFrag(A1, a1 + 32 * kc4, kof);
            acc0 = wmma_bf16(A0, WB[kc4], acc0);
            acc1 = wmma_bf16(A1, WB[kc4], acc1);
        }
        // D: VGPR r -> channel (chunk + r + 8g); N = lane&15 -> output column.
        if (gw < WW) {
#pragma unroll
            for (int r = 0; r < 8; ++r) {
                int ca = c0 + r + 8 * g;
                int cb = c0 + 16 + r + 8 * g;
                size_t gia = (((size_t)b * CC + ca) * HH + h) * WW + gw;
                size_t gib = (((size_t)b * CC + cb) * HH + h) * WW + gw;
                out[gia] = x[gia] + gamma * acc0[r];
                out[gib] = x[gib] + gamma * acc1[r];
            }
        }
    }
}

extern "C" void kernel_launch(void* const* d_in, const int* in_sizes, int n_in,
                              void* d_out, int out_size, void* d_ws, size_t ws_size,
                              hipStream_t stream)
{
    (void)in_sizes; (void)n_in; (void)out_size; (void)d_ws; (void)ws_size;
    const float* x     = (const float*)d_in[0];
    const float* gamma = (const float*)d_in[1];
    float*       out   = (float*)d_out;

    dim3 grid((WW + TW - 1) / TW, HH, BB);   // 4 x 56 x 4 = 896 blocks
    localppm_kernel<<<grid, NTH, 0, stream>>>(x, gamma, out);
}
